// ContextSNN_26963804684327
// MI455X (gfx1250) — matrix operand order repaired and checked by reading, not compiled, not run on hardware
//
#include <hip/hip_runtime.h>

// SNN forward: 50 timesteps x (GEMM+LIF)x3 on v_wmma_f32_16x16x32_f16.
// Weights f16 (L2-resident, ~46MB), spikes exact in f16, f32 accumulate.
// K-loop is software-pipelined with explicit double-buffered fragments so
// s_wait_loadcnt covers loads issued one iteration earlier (latency hiding
// inside each wave, on top of multi-wave occupancy).

typedef _Float16 v16h __attribute__((ext_vector_type(16)));
typedef _Float16 v8h  __attribute__((ext_vector_type(8)));
typedef float    v8f  __attribute__((ext_vector_type(8)));
typedef float    v4f  __attribute__((ext_vector_type(4)));

#define SNN_THR  1.0f
#define SNN_BETA 0.9f

__global__ void zero_f32(float* __restrict__ p, int n) {
  int i = blockIdx.x * blockDim.x + threadIdx.x;
  int stride = gridDim.x * blockDim.x;
  for (; i < n; i += stride) p[i] = 0.0f;
}

__global__ void f32_to_f16(const float* __restrict__ src, _Float16* __restrict__ dst, int n) {
  int i = blockIdx.x * blockDim.x + threadIdx.x;
  int stride = gridDim.x * blockDim.x;
  for (; i < n; i += stride) dst[i] = (_Float16)src[i];
}

// A-fragment loader, contiguous-k 16x32 f16 layout (B fragments use the same):
// lane(half,lm): row = lm, K range [kk+16*half, +16) contiguous.
// rowoff already includes row*K + 16*half.
template<bool A_F32>
__device__ __forceinline__ v16h load_a(const void* __restrict__ Ain, size_t rowoff, int kk) {
  if (A_F32) {
    const float* Ar = (const float*)Ain + rowoff + kk;
    v4f f0 = *(const v4f*)(Ar);
    v4f f1 = *(const v4f*)(Ar + 4);
    v4f f2 = *(const v4f*)(Ar + 8);
    v4f f3 = *(const v4f*)(Ar + 12);
    v16h a;
#pragma unroll
    for (int i = 0; i < 4; ++i) {
      a[i]      = (_Float16)f0[i];
      a[4 + i]  = (_Float16)f1[i];
      a[8 + i]  = (_Float16)f2[i];
      a[12 + i] = (_Float16)f3[i];
    }
    return a;
  } else {
    const _Float16* Ar = (const _Float16*)Ain + rowoff + kk;
    v8h lo = *(const v8h*)(Ar);
    v8h hi = *(const v8h*)(Ar + 8);
    return __builtin_shufflevector(lo, hi, 0,1,2,3,4,5,6,7,8,9,10,11,12,13,14,15);
  }
}

// Fused GEMM + LIF layer.
//   Ain : [M,K] spikes, f32 (layer 1) or f16 (layers 2/3), row-major
//   Wh  : [N,K] f16 weights row-major (== B^T: B columns K-contiguous, no transpose needed)
//   Mem : [M,N] f32 membrane; Out: [M,N] f16 spikes or f32 accumulator (+=)
// Block = 32*WVM*WVN threads. Wave (wv_m, wv_n) computes an (MT*16)x(NT*16) tile.
// Requires K % 64 == 0 (true for 1536 / 4096).
template<int MT, int NT, int WVM, int WVN, bool A_F32, bool ACC_OUT>
__global__ void __launch_bounds__(32 * WVM * WVN)
snn_layer(const void* __restrict__ Ain,
          const _Float16* __restrict__ Wh,
          const float* __restrict__ bias,
          float* __restrict__ Mem,
          void* __restrict__ Out,
          int M, int N, int K)
{
  const int lane = threadIdx.x & 31;
  const int wv   = threadIdx.x >> 5;
  const int half = lane >> 4;
  const int lm   = lane & 15;
  const int wv_m = wv % WVM;
  const int wv_n = wv / WVM;

  const int m0 = blockIdx.y * (WVM * MT * 16) + wv_m * (MT * 16);
  const int n0 = blockIdx.x * (WVN * NT * 16) + wv_n * (NT * 16);

  // Per-row base offsets/pointers (hoisted out of the K loop).
  size_t aOff[MT];
#pragma unroll
  for (int mt = 0; mt < MT; ++mt)
    aOff[mt] = (size_t)(m0 + mt * 16 + lm) * (size_t)K + 16 * half;

  const _Float16* pB[NT];
#pragma unroll
  for (int nt = 0; nt < NT; ++nt)
    pB[nt] = Wh + (size_t)(n0 + nt * 16 + lm) * (size_t)K + 16 * half;

  v8f acc[MT][NT] = {};

  // Double-buffered fragments: issue loads for the next k-step before the
  // WMMA chain of the current one.
  v16h a0[MT], a1[MT], b0[NT], b1[NT];
#pragma unroll
  for (int mt = 0; mt < MT; ++mt) a0[mt] = load_a<A_F32>(Ain, aOff[mt], 0);
#pragma unroll
  for (int nt = 0; nt < NT; ++nt) b0[nt] = *(const v16h*)(pB[nt]);

  for (int kk = 0; kk < K; kk += 64) {
    // prefetch k-step kk+32 (always in range: K % 64 == 0)
#pragma unroll
    for (int mt = 0; mt < MT; ++mt) a1[mt] = load_a<A_F32>(Ain, aOff[mt], kk + 32);
#pragma unroll
    for (int nt = 0; nt < NT; ++nt) b1[nt] = *(const v16h*)(pB[nt] + kk + 32);

#pragma unroll
    for (int nt = 0; nt < NT; ++nt)
#pragma unroll
      for (int mt = 0; mt < MT; ++mt)
        acc[mt][nt] = __builtin_amdgcn_wmma_f32_16x16x32_f16(
            false, a0[mt], false, b0[nt], (short)0, acc[mt][nt], false, false);

    // prefetch k-step kk+64 (guarded: last iteration has none)
    if (kk + 64 < K) {
#pragma unroll
      for (int mt = 0; mt < MT; ++mt) a0[mt] = load_a<A_F32>(Ain, aOff[mt], kk + 64);
#pragma unroll
      for (int nt = 0; nt < NT; ++nt) b0[nt] = *(const v16h*)(pB[nt] + kk + 64);
    }

#pragma unroll
    for (int nt = 0; nt < NT; ++nt)
#pragma unroll
      for (int mt = 0; mt < MT; ++mt)
        acc[mt][nt] = __builtin_amdgcn_wmma_f32_16x16x32_f16(
            false, a1[mt], false, b1[nt], (short)0, acc[mt][nt], false, false);
  }

  // Epilogue: bias + LIF (snntorch Leaky, reset-by-subtract, detached reset).
  // D layout: element r of acc[mt][nt] = row (m0+mt*16 + r + 8*half),
  //           col (n0+nt*16 + lm).
#pragma unroll
  for (int nt = 0; nt < NT; ++nt) {
    const int col = n0 + nt * 16 + lm;
    const float bn = bias[col];
#pragma unroll
    for (int mt = 0; mt < MT; ++mt) {
#pragma unroll
      for (int r = 0; r < 8; ++r) {
        const int row = m0 + mt * 16 + r + 8 * half;
        const size_t idx = (size_t)row * (size_t)N + col;
        const float cur   = acc[mt][nt][r] + bn;
        const float mo    = Mem[idx];
        const float reset = (mo > SNN_THR) ? SNN_THR : 0.0f;
        const float mn    = SNN_BETA * mo + cur - reset;
        Mem[idx] = mn;
        const float s = (mn > SNN_THR) ? 1.0f : 0.0f;   // spike = (mem - THR > 0)
        if (ACC_OUT) {
          ((float*)Out)[idx] += s;                      // rate accumulator
        } else {
          ((_Float16*)Out)[idx] = (_Float16)s;          // spikes for next layer
        }
      }
    }
  }
}

extern "C" void kernel_launch(void* const* d_in, const int* in_sizes, int n_in,
                              void* d_out, int out_size, void* d_ws, size_t ws_size,
                              hipStream_t stream) {
  constexpr int T = 50, Bt = 256, IN = 1536, H1 = 4096, H2 = 4096, OUT = 64;
  (void)in_sizes; (void)n_in; (void)out_size; (void)ws_size;

  const float* spikes = (const float*)d_in[0];  // [T,B,IN] f32 {0,1}
  const float* W1 = (const float*)d_in[1];      // [H1,IN]
  const float* b1 = (const float*)d_in[2];
  const float* W2 = (const float*)d_in[3];      // [H2,H1]
  const float* b2 = (const float*)d_in[4];
  const float* W3 = (const float*)d_in[5];      // [OUT,H2]
  const float* b3 = (const float*)d_in[6];
  float* out = (float*)d_out;                   // [B,OUT] f32

  // Workspace carve (~59 MB), 256B-aligned chunks; m1/m2/m3 contiguous.
  char* ws = (char*)d_ws;
  size_t off = 0;
  auto carve = [&](size_t bytes) -> char* {
    char* p = ws + off;
    off = (off + bytes + 255) & ~(size_t)255;
    return p;
  };
  _Float16* W1h = (_Float16*)carve((size_t)H1 * IN * 2);
  _Float16* W2h = (_Float16*)carve((size_t)H2 * H1 * 2);
  _Float16* W3h = (_Float16*)carve((size_t)OUT * H2 * 2);
  _Float16* s1  = (_Float16*)carve((size_t)Bt * H1 * 2);
  _Float16* s2  = (_Float16*)carve((size_t)Bt * H2 * 2);
  float* m1 = (float*)carve((size_t)Bt * H1 * 4);
  float* m2 = (float*)carve((size_t)Bt * H2 * 4);
  float* m3 = (float*)carve((size_t)Bt * OUT * 4);
  (void)m2; (void)m3;  // contiguous after m1; zeroed in one pass

  // Deterministic per-call init: membranes and output accumulator to zero.
  const int zn = Bt * H1 + Bt * H2 + Bt * OUT;
  zero_f32<<<256, 256, 0, stream>>>(m1, zn);
  zero_f32<<<64, 256, 0, stream>>>(out, Bt * OUT);

  // Weight conversion f32 -> f16.
  f32_to_f16<<<512, 256, 0, stream>>>(W1, W1h, H1 * IN);
  f32_to_f16<<<512, 256, 0, stream>>>(W2, W2h, H2 * H1);
  f32_to_f16<<<64,  256, 0, stream>>>(W3, W3h, OUT * H2);

  // Time loop: stream ordering enforces the layer/timestep dependency chain.
  // Wave tile 32x32 (MT=2,NT=2), block = 8 waves (2 along M, 4 along N)
  // -> block tile 64x128, 1024 waves for the H=4096 layers.
  for (int t = 0; t < T; ++t) {
    const float* xt = spikes + (size_t)t * Bt * IN;
    snn_layer<2, 2, 2, 4, true,  false>
        <<<dim3(H1 / 128, Bt / 64), 256, 0, stream>>>(xt, W1h, b1, m1, s1, Bt, H1, IN);
    snn_layer<2, 2, 2, 4, false, false>
        <<<dim3(H2 / 128, Bt / 64), 256, 0, stream>>>(s1, W2h, b2, m2, s2, Bt, H2, H1);
    snn_layer<2, 1, 2, 4, false, true>
        <<<dim3(OUT / 64, Bt / 64), 256, 0, stream>>>(s2, W3h, b3, m3, out, Bt, OUT, H2);
  }
}
